// RPerceptron_73452530696713
// MI455X (gfx1250) — matrix operand order repaired and checked by reading, not compiled, not run on hardware
//
#include <hip/hip_runtime.h>
#include <hip/hip_bf16.h>
#include <math.h>

typedef _Float16 half_t;
typedef __attribute__((ext_vector_type(16))) _Float16 v16h;
typedef __attribute__((ext_vector_type(8)))  _Float16 v8h;
typedef __attribute__((ext_vector_type(8)))  float    v8f;

#define B_SZ   32768
#define D_IN   512
#define M_SZ   2048
#define D_OUT  512
#define TOPK   8
#define ROWS_WG 32   // rows of x per workgroup (two 16-row WMMA tiles)

union V16U { v16h v; v8h h[2]; };

// ---------------------------------------------------------------------------
// L2-normalize each row of src (cols=512) and emit f16.
// ---------------------------------------------------------------------------
__global__ void __launch_bounds__(256) rownorm_f16(const float* __restrict__ src,
                                                   half_t* __restrict__ dst,
                                                   int cols) {
  const int row = blockIdx.x;
  const float* r = src + (size_t)row * cols;
  float ss = 0.0f;
  for (int c = threadIdx.x; c < cols; c += 256) { float v = r[c]; ss += v * v; }
#pragma unroll
  for (int off = 16; off; off >>= 1) ss += __shfl_xor(ss, off, 32);
  __shared__ float wsum[8];
  if ((threadIdx.x & 31) == 0) wsum[threadIdx.x >> 5] = ss;
  __syncthreads();
  float tot = 0.0f;
#pragma unroll
  for (int i = 0; i < 8; ++i) tot += wsum[i];
  const float scale = 1.0f / fmaxf(sqrtf(tot), 1e-12f);
  half_t* o = dst + (size_t)row * cols;
  for (int c = threadIdx.x; c < cols; c += 256) o[c] = (half_t)(r[c] * scale);
}

// ---------------------------------------------------------------------------
// Phase 1: S = x_h @ K_h^T via v_wmma_f32_16x16x32_f16.
// One workgroup = 32 rows x 2048 cols. 8 waves, each owns a 256-col stripe.
// Two resident A tiles (rows 0-15, 16-31) share each streamed B fragment,
// halving B (K_h) traffic vs the 16-row version. S slab (32*2048 f32 =
// 256 KB) lives in LDS; then per-row top-8 via shuffle-reduce argmax and a
// winner histogram via global f32 atomics.
// ---------------------------------------------------------------------------
__global__ void __launch_bounds__(256) phase1_gemm_topk(
    const half_t* __restrict__ xh, const half_t* __restrict__ kh,
    float* __restrict__ tv, int* __restrict__ ti, float* __restrict__ counts) {
  extern __shared__ float sS[];  // [32][2048]
  const int tid  = threadIdx.x;
  const int wave = tid >> 5;
  const int lane = tid & 31;
  const int g    = lane >> 4;    // half-wave group
  const int n    = lane & 15;    // row (A) / col (B,C) within fragment
  const int r0   = blockIdx.x * ROWS_WG;

  // A fragments, two 16-row tiles, all K (16 k-steps of 32), resident in VGPRs.
  // 16-bit A layout: lane group g holds halves K = k0+g*8+{0..7} and k0+16+g*8+{0..7}.
  V16U a0[16], a1[16];
  const half_t* ab0 = xh + (((size_t)(r0 + n))      << 9) + (g << 3);
  const half_t* ab1 = xh + (((size_t)(r0 + 16 + n)) << 9) + (g << 3);
#pragma unroll
  for (int t = 0; t < 16; ++t) {
    const half_t* p0 = ab0 + t * 32;
    const half_t* p1 = ab1 + t * 32;
    a0[t].h[0] = *(const v8h*)(p0);
    a0[t].h[1] = *(const v8h*)(p0 + 16);
    a1[t].h[0] = *(const v8h*)(p1);
    a1[t].h[1] = *(const v8h*)(p1 + 16);
  }

  for (int cb = 0; cb < 16; ++cb) {
    const int c0 = (wave << 8) + (cb << 4);
    v8f acc0 = {};
    v8f acc1 = {};
    // B fragment: col = c0+n (a K_norm row); lane group g holds K = k0+g*8+{0..7} and k0+16+g*8+{0..7}.
    const half_t* bb = kh + (((size_t)(c0 + n)) << 9) + (g << 3);
#pragma unroll
    for (int t = 0; t < 16; ++t) {
      V16U b;
      const half_t* q = bb + t * 32;
      b.h[0] = *(const v8h*)(q);
      b.h[1] = *(const v8h*)(q + 16);
      acc0 = __builtin_amdgcn_wmma_f32_16x16x32_f16(
          false, a0[t].v, false, b.v, (short)0, acc0, false, false);
      acc1 = __builtin_amdgcn_wmma_f32_16x16x32_f16(
          false, a1[t].v, false, b.v, (short)0, acc1, false, false);
    }
    // C/D layout: lane holds col n, rows v + 8*g across the 8 acc VGPRs.
#pragma unroll
    for (int v = 0; v < 8; ++v) {
      sS[((v + (g << 3))      << 11) + c0 + n] = acc0[v];
      sS[((v + (g << 3) + 16) << 11) + c0 + n] = acc1[v];
    }
  }
  __syncthreads();

  // Top-8 per row: wave w handles rows 4w .. 4w+3.
  for (int rr = 0; rr < 4; ++rr) {
    const int row = (wave << 2) + rr;
    float* Srow = sS + ((size_t)row << 11);
    const int gb = r0 + row;
    for (int it = 0; it < TOPK; ++it) {
      float bm = -INFINITY;
      int   bi = 0x7fffffff;
      for (int c = lane; c < M_SZ; c += 32) {
        float vv = Srow[c];
        if (vv > bm) { bm = vv; bi = c; }
      }
#pragma unroll
      for (int off = 16; off; off >>= 1) {
        float ov = __shfl_xor(bm, off, 32);
        int   oi = __shfl_xor(bi, off, 32);
        if (ov > bm || (ov == bm && oi < bi)) { bm = ov; bi = oi; }
      }
      if (lane == 0) {
        tv[(size_t)gb * TOPK + it] = bm;
        ti[(size_t)gb * TOPK + it] = bi;
        if (it == 0) atomicAdd(&counts[bi], 1.0f);  // argmax winner histogram
        Srow[bi] = -INFINITY;
      }
      __syncthreads();
    }
  }
}

// ---------------------------------------------------------------------------
// Phase 2: masked softmax over top-8 with log(s) + diversity bias, write
// attn row (zeros elsewhere), f, g, and y = g * sum_j w_j * V[idx_j].
// ---------------------------------------------------------------------------
__global__ void __launch_bounds__(256) phase2_out(
    const float* __restrict__ Vm, const float* __restrict__ s,
    const float* __restrict__ counts, const float* __restrict__ tv,
    const int* __restrict__ ti, float* __restrict__ y, float* __restrict__ f,
    float* __restrict__ g, float* __restrict__ attn) {
  const int b   = blockIdx.x;
  const int tid = threadIdx.x;

  float w[TOPK];
  int   id[TOPK];
  float m = -INFINITY;
#pragma unroll
  for (int j = 0; j < TOPK; ++j) {
    id[j] = ti[(size_t)b * TOPK + j];
    float sf = tv[(size_t)b * TOPK + j] * 10.0f          // /TAU
             + logf(s[id[j]] + 1e-8f)                    // log_s
             - 0.5f * counts[id[j]] * (1.0f / 32768.0f); // -GAMMA*counts/B
    w[j] = sf;
    m = fmaxf(m, sf);
  }
  float sum = 0.0f;
#pragma unroll
  for (int j = 0; j < TOPK; ++j) { w[j] = expf(w[j] - m); sum += w[j]; }
  const float inv = 1.0f / sum;
#pragma unroll
  for (int j = 0; j < TOPK; ++j) w[j] *= inv;

  const float fv = tv[(size_t)b * TOPK];                 // extraction is descending
  const float gv = 1.0f - 1.0f / (1.0f + expf(-10.0f * (fv - 0.5f)));
  if (tid == 0) { f[b] = fv; g[b] = gv; }

  float* arow = attn + ((size_t)b << 11);
  for (int c = tid; c < M_SZ; c += 256) arow[c] = 0.0f;
  __syncthreads();
  if (tid < TOPK) arow[id[tid]] = w[tid];

#pragma unroll
  for (int dd = 0; dd < 2; ++dd) {
    const int d = tid + dd * 256;
    float acc = 0.0f;
#pragma unroll
    for (int j = 0; j < TOPK; ++j) acc += w[j] * Vm[((size_t)id[j] << 9) + d];
    y[((size_t)b << 9) + d] = gv * acc;
  }
}

// ---------------------------------------------------------------------------
extern "C" void kernel_launch(void* const* d_in, const int* in_sizes, int n_in,
                              void* d_out, int out_size, void* d_ws, size_t ws_size,
                              hipStream_t stream) {
  const float* x = (const float*)d_in[0];
  const float* K = (const float*)d_in[1];
  const float* V = (const float*)d_in[2];
  const float* s = (const float*)d_in[3];

  float* out  = (float*)d_out;
  float* y    = out;                                   // B*512
  float* f    = out + (size_t)B_SZ * D_OUT;            // B
  float* g    = f + B_SZ;                              // B
  float* attn = g + B_SZ;                              // B*M

  char* ws = (char*)d_ws;
  const size_t XH = (size_t)B_SZ * D_IN * sizeof(half_t);   // 32 MB
  const size_t KH = (size_t)M_SZ * D_IN * sizeof(half_t);   // 2 MB
  half_t* xh     = (half_t*)(ws);
  half_t* kh     = (half_t*)(ws + XH);
  float*  counts = (float*)(ws + XH + KH);
  float*  tv     = (float*)(ws + XH + KH + M_SZ * sizeof(float));
  int*    ti     = (int*)(ws + XH + KH + M_SZ * sizeof(float)
                          + (size_t)B_SZ * TOPK * sizeof(float));

  rownorm_f16<<<B_SZ, 256, 0, stream>>>(x, xh, D_IN);
  rownorm_f16<<<M_SZ, 256, 0, stream>>>(K, kh, D_IN);
  hipMemsetAsync(counts, 0, M_SZ * sizeof(float), stream);
  phase1_gemm_topk<<<B_SZ / ROWS_WG, 256, ROWS_WG * M_SZ * sizeof(float), stream>>>(
      xh, kh, tv, ti, counts);
  phase2_out<<<B_SZ, 256, 0, stream>>>(V, s, counts, tv, ti, y, f, g, attn);
}
